// TrajectoryGeneratorAR_20675972563334
// MI455X (gfx1250) — matrix operand order repaired and checked by reading, not compiled, not run on hardware
//
#include <hip/hip_runtime.h>

#define SS    2048
#define NAG   16
#define OBSN  8
#define PREDN 12
#define HD    16
#define BTOT  (SS*NAG)
#define WPB   8            // waves (scenes) per 256-thread block

typedef __attribute__((ext_vector_type(16))) _Float16 v16h;
typedef __attribute__((ext_vector_type(8)))  float    v8f;

// Compiler-level fence: LDS ops within a wave are in-order in HW; this just
// stops clang from hoisting the fragment loads above the staging stores.
#define KFENCE() asm volatile("" ::: "memory")

#if defined(__has_builtin)
#if __has_builtin(__builtin_amdgcn_tanhf)
#define HAVE_HW_TANH 1
#endif
#endif

__device__ __forceinline__ float fast_tanh(float x){
#ifdef HAVE_HW_TANH
  return __builtin_amdgcn_tanhf(x);        // v_tanh_f32 (gfx1250 TRANS op)
#else
  const float e = __expf(2.0f * x);        // branch-free fallback
  return 1.0f - 2.0f / (e + 1.0f);
#endif
}

__device__ __forceinline__ float sigm(float x){
#ifdef HAVE_HW_TANH
  return 0.5f * (fast_tanh(0.5f * x) + 1.0f);   // one v_tanh_f32
#else
  return 1.0f / (1.0f + __expf(-x));
#endif
}

__device__ __forceinline__ v8f wmma_f16(v16h a, v16h b, v8f c){
  // D = A(16x32,f16) * B(32x16,f16) + C(16x16,f32)
  return __builtin_amdgcn_wmma_f32_16x16x32_f16(false, a, false, b, (short)0, c, false, false);
}

// Load A fragment (16x32 f16, row-major in LDS, row stride 32) into the contiguous-k A layout used here (same as B):
// lanes 0-15: M=lane, Ks 0..15; lanes 16-31: M=lane-16, Ks 16..31
__device__ __forceinline__ v16h load_A_tile(const _Float16* t, int lane){
  const int half = lane >> 4;
  const int m    = lane & 15;
  v16h a;
#pragma unroll
  for (int v = 0; v < 8; ++v){
    const int k0 = 16*half + 2*v;
    a[2*v]   = t[m*32 + k0];
    a[2*v+1] = t[m*32 + k0 + 1];
  }
  return a;
}

__global__ __launch_bounds__(256, 1) void traj_ar_kernel(
    const float* __restrict__ traj_rel, const float* __restrict__ obs_pos,
    const int*   __restrict__ nei,      const float* __restrict__ noise,
    const float* __restrict__ embW, const float* __restrict__ embB,
    const float* __restrict__ encWi, const float* __restrict__ encWh, const float* __restrict__ encB,
    const float* __restrict__ dinW, const float* __restrict__ dinB,
    const float* __restrict__ decWi, const float* __restrict__ decWh, const float* __restrict__ decB,
    const float* __restrict__ prW, const float* __restrict__ prB,
    const float* __restrict__ pmW, const float* __restrict__ pmB,
    const float* __restrict__ outW, const float* __restrict__ outB,
    float* __restrict__ out)
{
  __shared__ _Float16 sh_eh[WPB][16][32];   // [e | h] A staging (per wave)
  __shared__ _Float16 sh_pa[WPB][16][32];   // pool / dec_in A staging
  __shared__ float    sh_h  [WPB][16][16];  // h, natural layout
  __shared__ float    sh_ctx[WPB][16][16];  // pooled context, natural layout
  __shared__ float    sh_pos[WPB][16][2];
  __shared__ float    sh_o  [WPB][16][2];   // previous pred ("out" in carry)
  __shared__ unsigned sh_mask[WPB][16];     // neighbor bitmask per agent
  __shared__ float swEmbW[32], swEmbB[16];
  __shared__ float swPrW[32],  swPrB[16];
  __shared__ float swOutW[64], swOutB[4];

  const int tid = threadIdx.x;
  for (int i = tid; i < 32;  i += 256) swEmbW[i] = embW[i];
  for (int i = tid; i < 16;  i += 256) swEmbB[i] = embB[i];
  for (int i = tid; i < 32;  i += 256) swPrW[i]  = prW[i];
  for (int i = tid; i < 16;  i += 256) swPrB[i]  = prB[i];
  for (int i = tid; i < 64;  i += 256) swOutW[i] = outW[i];
  for (int i = tid; i < 4;   i += 256) swOutB[i] = outB[i];

  const int lane  = tid & 31;
  const int wv    = tid >> 5;
  const int scene = blockIdx.x * WPB + wv;
  const int half  = lane >> 4;
  const int col   = lane & 15;

  // ---- B fragments in registers, pre-swizzled (lane=column, K packed) ----
  // B layout assumed: lanes 0-15 hold K=0..15 of column `lane`; lanes 16-31 K=16..31.
  v16h BeA[4], BdA[4], Bp, Bdin;
#pragma unroll
  for (int w = 0; w < 4; ++w){
#pragma unroll
    for (int v = 0; v < 8; ++v){
#pragma unroll
      for (int u = 0; u < 2; ++u){
        const int k = 16*half + 2*v + u;                 // K index 0..31
        const float ev = (k < 16) ? encWi[k*64 + 16*w + col]
                                  : encWh[(k-16)*64 + 16*w + col];
        const float dv = (k < 16) ? decWi[k*64 + 16*w + col]
                                  : decWh[(k-16)*64 + 16*w + col];
        BeA[w][2*v+u] = (_Float16)ev;
        BdA[w][2*v+u] = (_Float16)dv;
      }
    }
  }
#pragma unroll
  for (int v = 0; v < 8; ++v)
#pragma unroll
    for (int u = 0; u < 2; ++u){
      const int k = 16*half + 2*v + u;
      Bp[2*v+u]   = (_Float16)pmW[k*16 + col];
      Bdin[2*v+u] = (k < 18) ? (_Float16)dinW[k*16 + col] : (_Float16)0.f;  // K=18 zero-padded
    }

  float bEnc[4], bDec[4];
#pragma unroll
  for (int w = 0; w < 4; ++w){ bEnc[w] = encB[16*w + col]; bDec[w] = decB[16*w + col]; }
  const float bPool = pmB[col];
  const float bDin  = dinB[col];

  if (lane < 16){
    unsigned m = 0u;
    const int base = (scene*NAG + lane) * NAG;
#pragma unroll
    for (int n = 0; n < NAG; ++n) if (nei[base + n] > 0) m |= (1u << n);
    sh_mask[wv][lane] = m;
  }

  float hreg[8], creg[8];                    // h, c in WMMA C layout
#pragma unroll
  for (int r = 0; r < 8; ++r){ hreg[r] = 0.f; creg[r] = 0.f; }

  __syncthreads();   // weights + masks visible

  // ======================= Encoder: 8 LSTM steps =======================
  for (int t = 0; t < OBSN; ++t){
#pragma unroll
    for (int i = 0; i < 8; ++i){             // e = relu(x @ embW + b), f16-staged
      const int idx = lane + 32*i;
      const int m = idx >> 4, j = idx & 15;
      const size_t p = ((size_t)t*BTOT + (size_t)scene*NAG + m)*2;
      const float e = fmaxf(traj_rel[p]*swEmbW[j] + traj_rel[p+1]*swEmbW[16+j] + swEmbB[j], 0.f);
      sh_eh[wv][m][j] = (_Float16)e;
    }
#pragma unroll
    for (int r = 0; r < 8; ++r)              // h into K=16..31 (C->natural store)
      sh_eh[wv][r + 8*half][16 + col] = (_Float16)hreg[r];
    KFENCE();
    const v16h a = load_A_tile(&sh_eh[wv][0][0], lane);
    v8f g[4];
#pragma unroll
    for (int w = 0; w < 4; ++w){
#pragma unroll
      for (int r = 0; r < 8; ++r) g[w][r] = bEnc[w];     // bias depends on col only
      g[w] = wmma_f16(a, BeA[w], g[w]);                  // 1 WMMA per gate
    }
#pragma unroll
    for (int r = 0; r < 8; ++r){
      const float ig = sigm(g[0][r]);
      const float fg = sigm(g[1][r]);
      const float gg = fast_tanh(g[2][r]);
      const float og = sigm(g[3][r]);
      creg[r] = fg*creg[r] + ig*gg;
      hreg[r] = og*fast_tanh(creg[r]);
    }
  }

  // ======================= Decoder init =======================
#pragma unroll
  for (int i = 0; i < 8; ++i){
    const int idx = lane + 32*i;
    sh_ctx[wv][idx >> 4][idx & 15] = 0.f;
  }
  {
    const int aa = lane >> 1, d = lane & 1;
    const size_t p = ((size_t)(OBSN-1)*BTOT + (size_t)scene*NAG + aa)*2 + d;
    sh_pos[wv][aa][d] = obs_pos[p];
    sh_o[wv][aa][d]   = traj_rel[p];
  }
  KFENCE();

  const size_t offMu  = (size_t)PREDN*BTOT*2;
  const size_t offStd = 2*(size_t)PREDN*BTOT*2;

  // ======================= Decoder: 12 steps =======================
  for (int t = 0; t < PREDN; ++t){
    // ---- dec_in as WMMA: e = relu([ctx | out | 0](16x32) @ Bdin + b) ----
#pragma unroll
    for (int i = 0; i < 8; ++i){
      const int idx = lane + 32*i;
      const int m = idx >> 4, j = idx & 15;
      sh_pa[wv][m][j]      = (_Float16)sh_ctx[wv][m][j];                     // K 0..15 : ctx
      sh_pa[wv][m][16 + j] = (j < 2) ? (_Float16)sh_o[wv][m][j] : (_Float16)0.f; // K 16,17 : out
    }
    KFENCE();
    {
      const v16h a2 = load_A_tile(&sh_pa[wv][0][0], lane);
      v8f eacc;
#pragma unroll
      for (int r = 0; r < 8; ++r) eacc[r] = bDin;
      eacc = wmma_f16(a2, Bdin, eacc);
#pragma unroll
      for (int r = 0; r < 8; ++r)            // e (C layout) -> sh_eh K=0..15
        sh_eh[wv][r + 8*half][col] = (_Float16)fmaxf(eacc[r], 0.f);
    }
#pragma unroll
    for (int r = 0; r < 8; ++r)              // h -> sh_eh K=16..31
      sh_eh[wv][r + 8*half][16 + col] = (_Float16)hreg[r];
    KFENCE();
    {
      const v16h a = load_A_tile(&sh_eh[wv][0][0], lane);
      v8f g[4];
#pragma unroll
      for (int w = 0; w < 4; ++w){
#pragma unroll
        for (int r = 0; r < 8; ++r) g[w][r] = bDec[w];
        g[w] = wmma_f16(a, BdA[w], g[w]);
      }
#pragma unroll
      for (int r = 0; r < 8; ++r){
        const float ig = sigm(g[0][r]);
        const float fg = sigm(g[1][r]);
        const float gg = fast_tanh(g[2][r]);
        const float og = sigm(g[3][r]);
        creg[r] = fg*creg[r] + ig*gg;
        hreg[r] = og*fast_tanh(creg[r]);
      }
    }
#pragma unroll
    for (int r = 0; r < 8; ++r)              // publish new h (natural layout)
      sh_h[wv][r + 8*half][col] = hreg[r];
    KFENCE();

    // ---- social pooling: 16 M-tiles, one WMMA each ----
    for (int at = 0; at < NAG; ++at){
      const float pax = sh_pos[wv][at][0], pay = sh_pos[wv][at][1];
#pragma unroll
      for (int i = 0; i < 8; ++i){
        const int idx = lane + 32*i;
        const int n = idx >> 4, j = idx & 15;
        const float rx = pax - sh_pos[wv][n][0];
        const float ry = pay - sh_pos[wv][n][1];
        const float re = fmaxf(rx*swPrW[j] + ry*swPrW[16+j] + swPrB[j], 0.f);
        sh_pa[wv][n][j]      = (_Float16)re;               // K 0..15  : rel-embed
        sh_pa[wv][n][16 + j] = (_Float16)sh_h[wv][n][j];   // K 16..31 : h[neighbor]
      }
      KFENCE();
      const v16h ap = load_A_tile(&sh_pa[wv][0][0], lane);
      v8f f;
#pragma unroll
      for (int r = 0; r < 8; ++r) f[r] = bPool;
      f = wmma_f16(ap, Bp, f);
      const unsigned mb = sh_mask[wv][at];
      float mx = -1.0e9f;
#pragma unroll
      for (int r = 0; r < 8; ++r){            // masked max over neighbor rows
        const int n = r + 8*half;
        const float v = fmaxf(f[r], 0.f);     // relu then mask
        mx = fmaxf(mx, ((mb >> n) & 1u) ? v : -1.0e9f);
      }
      mx = fmaxf(mx, __shfl_xor(mx, 16, 32)); // combine lane halves (rows 0-7 / 8-15)
      const float pooled = (mb != 0u) ? mx : 0.f;
      if (lane < 16) sh_ctx[wv][at][lane] = pooled;
    }
    KFENCE();

    // ---- output head: o4 = (h + ctx) @ outW + b; pred = mu + std*noise ----
    if (lane < 16){
      const int aa = lane;
      float o4[4];
#pragma unroll
      for (int j = 0; j < 4; ++j){
        float s = swOutB[j];
#pragma unroll
        for (int k = 0; k < 16; ++k)
          s += (sh_h[wv][aa][k] + sh_ctx[wv][aa][k]) * swOutW[k*4 + j];
        o4[j] = s;
      }
      const float mux = o4[0], muy = o4[1];
      const float sdx = __expf(fminf(fmaxf(o4[2], -9.f), 4.f));
      const float sdy = __expf(fminf(fmaxf(o4[3], -9.f), 4.f));
      const size_t b  = (size_t)scene*NAG + aa;
      const float nx = noise[((size_t)t*BTOT + b)*2 + 0];
      const float ny = noise[((size_t)t*BTOT + b)*2 + 1];
      const float px = mux + sdx*nx, py = muy + sdy*ny;
      const size_t e2 = ((size_t)t*BTOT + b)*2;
      out[e2 + 0]          = px;  out[e2 + 1]          = py;
      out[offMu  + e2 + 0] = mux; out[offMu  + e2 + 1] = muy;
      out[offStd + e2 + 0] = sdx; out[offStd + e2 + 1] = sdy;
      sh_o[wv][aa][0] = px;  sh_o[wv][aa][1] = py;
      sh_pos[wv][aa][0] += px; sh_pos[wv][aa][1] += py;
    }
    KFENCE();
  }
}

extern "C" void kernel_launch(void* const* d_in, const int* in_sizes, int n_in,
                              void* d_out, int out_size, void* d_ws, size_t ws_size,
                              hipStream_t stream) {
  (void)in_sizes; (void)n_in; (void)out_size; (void)d_ws; (void)ws_size;
  traj_ar_kernel<<<SS/WPB, 256, 0, stream>>>(
      (const float*)d_in[0],  (const float*)d_in[1],
      (const int*  )d_in[2],  (const float*)d_in[3],
      (const float*)d_in[4],  (const float*)d_in[5],
      (const float*)d_in[6],  (const float*)d_in[7],  (const float*)d_in[8],
      (const float*)d_in[9],  (const float*)d_in[10],
      (const float*)d_in[11], (const float*)d_in[12], (const float*)d_in[13],
      (const float*)d_in[14], (const float*)d_in[15],
      (const float*)d_in[16], (const float*)d_in[17],
      (const float*)d_in[18], (const float*)d_in[19],
      (float*)d_out);
}
